// Decoder_40046275068010
// MI455X (gfx1250) — matrix operand order repaired and checked by reading, not compiled, not run on hardware
//
#include <hip/hip_runtime.h>
#include <hip/hip_bf16.h>

// ---------------- problem constants ----------------
#define NN   12000
#define EE   384000
#define EP   (EE + NN)        // 396000 edges incl. self loops
#define FIN  4
#define FH   50
#define FO   1433
#define FOP  1440             // FO padded to multiple of 32 for WMMA K-loop
#define NEG_SLOPE 0.2f

typedef _Float16 v8h  __attribute__((ext_vector_type(8)));
typedef _Float16 v16h __attribute__((ext_vector_type(16)));
typedef float    v8f  __attribute__((ext_vector_type(8)));

// ---------------- helpers ----------------
__device__ __forceinline__ void edge_nodes(const int* __restrict__ ei, int idx,
                                           int& sN, int& dN) {
  if (idx < EE) { sN = ei[idx]; dN = ei[EE + idx]; }
  else          { sN = idx - EE; dN = idx - EE; }   // appended self loops
}

__device__ __forceinline__ void atomicMaxFloat(float* addr, float v) {
  // standard sign-split trick; m initialized to -inf
  if (v >= 0.f) atomicMax((int*)addr, __float_as_int(v));
  else          atomicMin((unsigned int*)addr, __float_as_uint(v));
}

__device__ __forceinline__ float leaky(float e) {
  return e > 0.f ? e : NEG_SLOPE * e;
}

// ---------------- layer 1: tiny dense transform + logit parts ----------------
__global__ void k_linear1(const float* __restrict__ x, const float* __restrict__ W1,
                          const float* __restrict__ as1, const float* __restrict__ ad1,
                          float* __restrict__ h1, float* __restrict__ al,
                          float* __restrict__ ar) {
  int n = blockIdx.x * blockDim.x + threadIdx.x;
  if (n >= NN) return;
  float xi[FIN];
#pragma unroll
  for (int i = 0; i < FIN; ++i) xi[i] = x[n * FIN + i];
  float sal = 0.f, sar = 0.f;
  for (int j = 0; j < FH; ++j) {
    float v = 0.f;
#pragma unroll
    for (int i = 0; i < FIN; ++i) v = fmaf(xi[i], W1[i * FH + j], v);
    h1[n * FH + j] = v;
    sal = fmaf(v, as1[j], sal);
    sar = fmaf(v, ad1[j], sar);
  }
  al[n] = sal; ar[n] = sar;
}

// ---------------- init kernels ----------------
__global__ void k_init1(float* __restrict__ out1, float* __restrict__ m,
                        float* __restrict__ s) {
  int i = blockIdx.x * blockDim.x + threadIdx.x;
  if (i < NN * FH) out1[i] = 0.f;
  if (i < NN) { m[i] = -__builtin_inff(); s[i] = 0.f; }
}

__global__ void k_zero(float* __restrict__ p, long cnt) {
  long i = (long)blockIdx.x * blockDim.x + threadIdx.x;
  long stride = (long)gridDim.x * blockDim.x;
  for (; i < cnt; i += stride) p[i] = 0.f;
}

__global__ void k_initms(float* __restrict__ m, float* __restrict__ s) {
  int i = blockIdx.x * blockDim.x + threadIdx.x;
  if (i < NN) { m[i] = -__builtin_inff(); s[i] = 0.f; }
}

// ---------------- edge softmax passes (shared by both layers) ----------------
__global__ void k_edge_max(const float* __restrict__ al, const float* __restrict__ ar,
                           float* __restrict__ m, const int* __restrict__ ei) {
  int i = blockIdx.x * blockDim.x + threadIdx.x;
  if (i >= EP) return;
  int sN, dN; edge_nodes(ei, i, sN, dN);
  atomicMaxFloat(&m[dN], leaky(al[sN] + ar[dN]));
}

__global__ void k_edge_exp(const float* __restrict__ al, const float* __restrict__ ar,
                           const float* __restrict__ m, float* __restrict__ s,
                           float* __restrict__ ex, const int* __restrict__ ei) {
  int i = blockIdx.x * blockDim.x + threadIdx.x;
  if (i >= EP) return;
  int sN, dN; edge_nodes(ei, i, sN, dN);
  float e = leaky(al[sN] + ar[dN]);
  float p = expf(e - m[dN]);
  ex[i] = p;
  atomicAdd(&s[dN], p);
}

// layer-1 aggregation: one thread per (edge, feature)
__global__ void k_agg1(const float* __restrict__ ex, const float* __restrict__ s,
                       const float* __restrict__ h1, float* __restrict__ out1,
                       const int* __restrict__ ei) {
  long idx = (long)blockIdx.x * blockDim.x + threadIdx.x;
  if (idx >= (long)EP * FH) return;
  int e = (int)(idx / FH), f = (int)(idx - (long)e * FH);
  int sN, dN; edge_nodes(ei, e, sN, dN);
  float alpha = ex[e] / s[dN];
  atomicAdd(&out1[dN * FH + f], alpha * h1[sN * FH + f]);
}

__global__ void k_relu1(const float* __restrict__ out1, const float* __restrict__ b1,
                        float* __restrict__ x2) {
  int i = blockIdx.x * blockDim.x + threadIdx.x;
  if (i >= NN * FH) return;
  float v = out1[i] + b1[i % FH];
  x2[i] = v > 0.f ? v : 0.f;
}

// ---------------- layer 2: dense transform [N,50] x [50,1433] ----------------
__global__ void k_linear2(const float* __restrict__ x2, const float* __restrict__ W2,
                          float* __restrict__ h2) {
  long idx = (long)blockIdx.x * blockDim.x + threadIdx.x;
  if (idx >= (long)NN * FO) return;
  int n = (int)(idx / FO), o = (int)(idx - (long)n * FO);
  float v = 0.f;
  const float* xr = x2 + n * FH;
#pragma unroll 10
  for (int k = 0; k < FH; ++k) v = fmaf(xr[k], W2[k * FO + o], v);
  h2[idx] = v;
}

// per-node dot with attention vectors, LDS tree reduction
__global__ void k_dots2(const float* __restrict__ h2, const float* __restrict__ as2,
                        const float* __restrict__ ad2, float* __restrict__ al,
                        float* __restrict__ ar) {
  __shared__ float s1[256], s2[256];
  int n = blockIdx.x, t = threadIdx.x;
  float a = 0.f, b = 0.f;
  for (int o = t; o < FO; o += 256) {
    float v = h2[(long)n * FO + o];
    a = fmaf(v, as2[o], a);
    b = fmaf(v, ad2[o], b);
  }
  s1[t] = a; s2[t] = b;
  __syncthreads();
  for (int st = 128; st > 0; st >>= 1) {
    if (t < st) { s1[t] += s1[t + st]; s2[t] += s2[t + st]; }
    __syncthreads();
  }
  if (t == 0) { al[n] = s1[0]; ar[n] = s2[0]; }
}

// layer-2 aggregation: one block per edge, threads stride features
__global__ void k_agg2(const float* __restrict__ ex, const float* __restrict__ s,
                       const float* __restrict__ h2, float* __restrict__ out2,
                       const int* __restrict__ ei) {
  int e = blockIdx.x;
  int sN, dN; edge_nodes(ei, e, sN, dN);
  float alpha = ex[e] / s[dN];
  const float* hs = h2 + (long)sN * FO;
  float* od = out2 + (long)dN * FO;
  for (int f = threadIdx.x; f < FO; f += blockDim.x)
    atomicAdd(&od[f], alpha * hs[f]);
}

// relu + bias + convert to f16 with zero padding to FOP columns
__global__ void k_pack(const float* __restrict__ out2, const float* __restrict__ b2,
                       _Float16* __restrict__ hf) {
  long idx = (long)blockIdx.x * blockDim.x + threadIdx.x;
  if (idx >= (long)NN * FOP) return;
  int n = (int)(idx / FOP), f = (int)(idx - (long)n * FOP);
  float v = 0.f;
  if (f < FO) {
    v = out2[(long)n * FO + f] + b2[f];
    v = v > 0.f ? v : 0.f;
  }
  hf[idx] = (_Float16)v;
}

// ---------------- decoder: sigmoid(hf @ hf^T) via WMMA f16 ----------------
// block = 128 threads (4 waves in 2x2), wave tile 48x48 (3x3 WMMA accs),
// block tile 96x96, grid 125x125 (12000 = 125*96 exactly -> EXEC all ones).
__global__ void __launch_bounds__(128) k_gemm_sig(const _Float16* __restrict__ hf,
                                                  float* __restrict__ out) {
  const int lane = threadIdx.x & 31;
  const int wave = threadIdx.x >> 5;
  const int wr = wave >> 1, wc = wave & 1;
  const int m0 = blockIdx.x * 96 + wr * 48;
  const int n0 = blockIdx.y * 96 + wc * 48;
  const bool hi = lane >= 16;
  const int l15 = lane & 15;
  const int aoff = hi ? 8 : 0;    // A: lanes 16-31 start at K=8 / K=24
  const int boff = hi ? 16 : 0;   // unused: B is loaded with aoff, same K as A

  const _Float16* arow[3];
  const _Float16* bcol[3];
#pragma unroll
  for (int t = 0; t < 3; ++t) {
    arow[t] = hf + (long)(m0 + 16 * t + l15) * FOP;   // A row = output row
    bcol[t] = hf + (long)(n0 + 16 * t + l15) * FOP;   // B col = h row (B = h^T)
  }

  v8f acc[3][3];
#pragma unroll
  for (int i = 0; i < 3; ++i)
#pragma unroll
    for (int j = 0; j < 3; ++j) acc[i][j] = (v8f){};

  for (int k0 = 0; k0 < FOP; k0 += 32) {
    v16h A[3], B[3];
#pragma unroll
    for (int t = 0; t < 3; ++t) {
      v8h lo = *(const v8h*)(arow[t] + k0 + aoff);
      v8h hh = *(const v8h*)(arow[t] + k0 + 16 + aoff);
      A[t] = __builtin_shufflevector(lo, hh, 0, 1, 2, 3, 4, 5, 6, 7,
                                              8, 9, 10, 11, 12, 13, 14, 15);
      B[t] = __builtin_shufflevector(*(const v8h*)(bcol[t] + k0 + aoff),
                                     *(const v8h*)(bcol[t] + k0 + 16 + aoff),
                                     0, 1, 2, 3, 4, 5, 6, 7,
                                     8, 9, 10, 11, 12, 13, 14, 15);
    }
    if (k0 + 64 < FOP) {                // hint next chunk into cache
      __builtin_prefetch(arow[0] + k0 + 64, 0, 0);
      __builtin_prefetch(bcol[0] + k0 + 64, 0, 0);
    }
#pragma unroll
    for (int i = 0; i < 3; ++i)
#pragma unroll
      for (int j = 0; j < 3; ++j)
        acc[i][j] = __builtin_amdgcn_wmma_f32_16x16x32_f16(
            false, A[i], false, B[j], (short)0, acc[i][j], false, false);
  }

  // epilogue: C/D layout -> global, fused sigmoid
#pragma unroll
  for (int i = 0; i < 3; ++i) {
    int rbase = m0 + 16 * i + (hi ? 8 : 0);
#pragma unroll
    for (int j = 0; j < 3; ++j) {
      int col = n0 + 16 * j + l15;
#pragma unroll
      for (int r = 0; r < 8; ++r) {
        float v = acc[i][j][r];
        out[(long)(rbase + r) * NN + col] = 1.f / (1.f + expf(-v));
      }
    }
  }
}

// copy edge_index (second tuple element) into the output tail
__global__ void k_copy_edges(const int* __restrict__ ei, int* __restrict__ tail) {
  int i = blockIdx.x * blockDim.x + threadIdx.x;
  if (i < 2 * EE) tail[i] = ei[i];
}

// ---------------- host-side orchestration ----------------
extern "C" void kernel_launch(void* const* d_in, const int* in_sizes, int n_in,
                              void* d_out, int out_size, void* d_ws, size_t ws_size,
                              hipStream_t stream) {
  const float* x   = (const float*)d_in[0];
  const int*   ei  = (const int*)d_in[1];
  const float* W1  = (const float*)d_in[2];
  const float* as1 = (const float*)d_in[3];
  const float* ad1 = (const float*)d_in[4];
  const float* b1  = (const float*)d_in[5];
  const float* W2  = (const float*)d_in[6];
  const float* as2 = (const float*)d_in[7];
  const float* ad2 = (const float*)d_in[8];
  const float* b2  = (const float*)d_in[9];

  // workspace layout (all 256B aligned)
  char* w = (char*)d_ws;
  size_t off = 0;
  auto alloc = [&](size_t bytes) {
    void* p = w + off;
    off += (bytes + 255) & ~(size_t)255;
    return p;
  };
  float* h1   = (float*)alloc((size_t)NN * FH * 4);
  float* x2   = (float*)alloc((size_t)NN * FH * 4);
  float* out1 = (float*)alloc((size_t)NN * FH * 4);
  float* al   = (float*)alloc((size_t)NN * 4);
  float* ar   = (float*)alloc((size_t)NN * 4);
  float* m    = (float*)alloc((size_t)NN * 4);
  float* s    = (float*)alloc((size_t)NN * 4);
  float* ex   = (float*)alloc((size_t)EP * 4);
  _Float16* hf = (_Float16*)alloc((size_t)NN * FOP * 2);
  (void)ws_size; (void)n_in; (void)in_sizes; (void)out_size;

  // large intermediates staged inside d_out (576MB); final GEMM overwrites them
  float* ob   = (float*)d_out;
  float* h2   = ob;                    // N*FO = 17.2M floats
  float* out2 = ob + 20000000;         // well past h2, well before 144M

  const int T = 256;

  // ---- layer 1 ----
  k_linear1<<<(NN + 127) / 128, 128, 0, stream>>>(x, W1, as1, ad1, h1, al, ar);
  k_init1<<<(NN * FH + T - 1) / T, T, 0, stream>>>(out1, m, s);
  k_edge_max<<<(EP + T - 1) / T, T, 0, stream>>>(al, ar, m, ei);
  k_edge_exp<<<(EP + T - 1) / T, T, 0, stream>>>(al, ar, m, s, ex, ei);
  {
    long tot = (long)EP * FH;
    k_agg1<<<(unsigned)((tot + T - 1) / T), T, 0, stream>>>(ex, s, h1, out1, ei);
  }
  k_relu1<<<(NN * FH + T - 1) / T, T, 0, stream>>>(out1, b1, x2);

  // ---- layer 2 ----
  {
    long tot = (long)NN * FO;
    k_linear2<<<(unsigned)((tot + T - 1) / T), T, 0, stream>>>(x2, W2, h2);
  }
  k_dots2<<<NN, 256, 0, stream>>>(h2, as2, ad2, al, ar);
  k_zero<<<4096, T, 0, stream>>>(out2, (long)NN * FO);
  k_initms<<<(NN + T - 1) / T, T, 0, stream>>>(m, s);
  k_edge_max<<<(EP + T - 1) / T, T, 0, stream>>>(al, ar, m, ei);
  k_edge_exp<<<(EP + T - 1) / T, T, 0, stream>>>(al, ar, m, s, ex, ei);
  k_agg2<<<EP, 256, 0, stream>>>(ex, s, h2, out2, ei);
  {
    long tot = (long)NN * FOP;
    k_pack<<<(unsigned)((tot + T - 1) / T), T, 0, stream>>>(out2, b2, hf);
  }

  // ---- decoder: sigmoid(h h^T), WMMA f16 ----
  dim3 g(125, 125);
  k_gemm_sig<<<g, 128, 0, stream>>>(hf, ob);

  // ---- tuple element 2: edge_index passthrough ----
  int* tail = (int*)(ob + (size_t)NN * NN);
  k_copy_edges<<<(2 * EE + T - 1) / T, T, 0, stream>>>(ei, tail);
}
